// SystemRobots_31568009625616
// MI455X (gfx1250) — compile-verified, matrix operand order checked by reading, not run on hardware
//
#include <hip/hip_runtime.h>
#include <hip/hip_bf16.h>

typedef __attribute__((ext_vector_type(16))) _Float16 v16h;
typedef __attribute__((ext_vector_type(8)))  _Float16 v8h;
typedef __attribute__((ext_vector_type(8)))  float    v8f;

#define BATCH       512
#define TSTEPS      200
#define SDIM        512   // 4 * n_agents
#define UDIM        256   // 2 * n_agents
#define ROWS_PER_WG 16

static __device__ __forceinline__ v16h combine16(v8h lo, v8h hi) {
    return __builtin_shufflevector(lo, hi, 0,1,2,3,4,5,6,7,8,9,10,11,12,13,14,15);
}

// one-time: K (f32, row-major [UDIM][SDIM]) -> f16 copy in workspace
__global__ void k_convert_K(const float* __restrict__ K, _Float16* __restrict__ Kh) {
    int i = blockIdx.x * blockDim.x + threadIdx.x;
    if (i < UDIM * SDIM) Kh[i] = (_Float16)K[i];
}

// Persistent scan kernel: each WG owns 16 batch rows for all 200 steps.
// 512 threads = 16 wave32 waves; wave w computes u columns [16w, 16w+16).
__global__ __launch_bounds__(512)
void k_scan(const float* __restrict__ data, const float* __restrict__ xbar,
            const _Float16* __restrict__ Kh,
            float* __restrict__ xlog, float* __restrict__ ulog)
{
    __shared__ __align__(16) float    s_x [ROWS_PER_WG][SDIM];   // 32 KB f32 state
    __shared__ __align__(16) _Float16 s_xh[ROWS_PER_WG][SDIM];   // 16 KB f16 WMMA-A mirror
    __shared__ __align__(16) float    s_u [ROWS_PER_WG][UDIM];   // 16 KB control state

    const int tid  = threadIdx.x;
    const int row  = tid >> 5;          // batch row within WG (== wave id)
    const int lane = tid & 31;
    const int half = lane >> 4;         // lane half selects K-subgroup in A/B frags
    const int l16  = lane & 15;
    const int r0   = blockIdx.x * ROWS_PER_WG;

    const float h = 0.05f;              // step
    const float c = -0.005f;            // h * (-b2/mass)

    // xbar slice for this lane's 16 consecutive state entries (same for all rows)
    float xb[16];
    #pragma unroll
    for (int j = 0; j < 4; ++j) {
        const float4 t4 = ((const float4*)xbar)[lane * 4 + j];
        xb[4*j+0] = t4.x; xb[4*j+1] = t4.y; xb[4*j+2] = t4.z; xb[4*j+3] = t4.w;
    }

    // Register-resident B fragments: K^T tile (columns [nb, nb+16)), all 16 k-steps.
    // Per-lane layout mirrors the 16-bit A layout: elements 0-7 <- k = 32ks+8*half..+7,
    // elements 8-15 <- k = 32ks+16+8*half..+7  (two contiguous 16-byte runs).
    const int nb   = row * 16;                 // n-tile base for this wave
    const int ncol = nb + l16;                 // output column (K row index)
    v16h bfr[16];
    #pragma unroll
    for (int ks = 0; ks < 16; ++ks) {
        const _Float16* p = Kh + (size_t)ncol * SDIM + 32 * ks + 8 * half;
        bfr[ks] = combine16(*(const v8h*)p, *(const v8h*)(p + 16));
    }

    // init recurrence: x = xbar, u = 0
    #pragma unroll
    for (int j = 0; j < 16; ++j) s_x[row][lane * 16 + j] = xb[j];
    #pragma unroll
    for (int j = 0; j < 8;  ++j) s_u[row][lane * 8 + j] = 0.0f;
    __syncthreads();

    for (int t = 0; t < TSTEPS; ++t) {
        // ---- phase A: block-diagonal dynamics, elementwise per agent ----
        const float* wrow = data + ((size_t)(r0 + row) * TSTEPS + t) * SDIM + lane * 16;
        float*       xrow = xlog + ((size_t)(r0 + row) * TSTEPS + t) * SDIM + lane * 16;
        #pragma unroll
        for (int a = 0; a < 4; ++a) {
            const int s  = lane * 16 + 4 * a;   // state offset of agent block
            const int ag = lane * 4 + a;        // agent index
            float p0 = s_x[row][s+0], p1 = s_x[row][s+1];
            float v0 = s_x[row][s+2], v1 = s_x[row][s+3];
            float u0 = s_u[row][2*ag+0], u1 = s_u[row][2*ag+1];
            const float4 w4 = *(const float4*)(wrow + 4 * a);
            const float xb0 = xb[4*a+0], xb1 = xb[4*a+1], xb2 = xb[4*a+2], xb3 = xb[4*a+3];
            float dp0 = p0 - xb0, dp1 = p1 - xb1, dv0 = v0 - xb2, dv1 = v1 - xb3;
            float sp  = sqrtf(v0 * v0 + v1 * v1);          // speed_diag uses raw x velocity
            float np0 = dp0 + h * dv0 + xb0 + w4.x;
            float np1 = dp1 + h * dv1 + xb1 + w4.y;
            float nv0 = dv0 + h * (-dp0 - dv0) + c * sp * dv0 + h * u0 + xb2 + w4.z;
            float nv1 = dv1 + h * (-dp1 - dv1) + c * sp * dv1 + h * u1 + xb3 + w4.w;
            s_x[row][s+0] = np0; s_x[row][s+1] = np1;
            s_x[row][s+2] = nv0; s_x[row][s+3] = nv1;
            s_xh[row][s+0] = (_Float16)np0; s_xh[row][s+1] = (_Float16)np1;
            s_xh[row][s+2] = (_Float16)nv0; s_xh[row][s+3] = (_Float16)nv1;
            *(float4*)(xrow + 4 * a) = make_float4(np0, np1, nv0, nv1);
        }
        __syncthreads();   // xh ready; u fully consumed

        // ---- phase B: u_new = x_new @ K^T, one 16x16 n-tile per wave, K=512 in 16 WMMAs ----
        v8f acc = {};
        #pragma unroll
        for (int ks = 0; ks < 16; ++ks) {
            const _Float16* ap = &s_xh[l16][32 * ks + 8 * half];   // A: lane%16 = M row
            v16h af = combine16(*(const v8h*)ap, *(const v8h*)(ap + 16));
            acc = __builtin_amdgcn_wmma_f32_16x16x32_f16(
                      /*neg_a=*/false, af, /*neg_b=*/false, bfr[ks],
                      /*c_mod=*/(short)0, acc, /*reuse_a=*/false, /*reuse_b=*/false);
        }
        // D layout: VGPR d -> row d + 8*half, column lane%16
        #pragma unroll
        for (int d = 0; d < 8; ++d) {
            const int   mr  = d + 8 * half;
            const float val = acc[d];
            s_u[mr][nb + l16] = val;
            ulog[((size_t)(r0 + mr) * TSTEPS + t) * UDIM + nb + l16] = val;
        }
        __syncthreads();   // u ready for next step's dynamics
    }
}

extern "C" void kernel_launch(void* const* d_in, const int* in_sizes, int n_in,
                              void* d_out, int out_size, void* d_ws, size_t ws_size,
                              hipStream_t stream) {
    (void)in_sizes; (void)n_in; (void)out_size; (void)ws_size;
    const float* data = (const float*)d_in[0];   // [512, 200, 512] f32
    const float* xbar = (const float*)d_in[1];   // [512] f32
    const float* K    = (const float*)d_in[2];   // [256, 512] f32
    _Float16* Kh = (_Float16*)d_ws;              // 256 KB f16 copy of K

    float* xlog = (float*)d_out;                                   // [512, 200, 512]
    float* ulog = xlog + (size_t)BATCH * TSTEPS * SDIM;            // [512, 200, 256]

    k_convert_K<<<(UDIM * SDIM + 255) / 256, 256, 0, stream>>>(K, Kh);
    k_scan<<<BATCH / ROWS_PER_WG, 512, 0, stream>>>(data, xbar, Kh, xlog, ulog);
}
